// CosineAttention_68985764708715
// MI455X (gfx1250) — matrix operand order repaired and checked by reading, not compiled, not run on hardware
//
#include <hip/hip_runtime.h>
#include <hip/hip_fp16.h>
#include <math.h>

typedef __attribute__((ext_vector_type(16))) _Float16 v16h;
typedef __attribute__((ext_vector_type(8)))  float    v8f;
typedef __attribute__((ext_vector_type(2)))  __fp16   fp16x2;

static constexpr int   CIN  = 256;   // input channels
static constexpr int   PIX  = 4096;  // 64*64
static constexpr int   NH   = 8;
static constexpr int   HD   = 32;    // head_dim == WMMA K
static constexpr float EPSN = 1e-4f;
// Q folded scale: (1/sqrt(32)) * log2(e); logits capped by sqrt(32)*log2(e)
static constexpr float QSCALE = 0.25503583f;
static constexpr float LOGCAP = 8.161116f;

union HF16 { v16h v; unsigned u[8]; uint4 q4[2]; _Float16 h[16]; };

// ---------------------------------------------------------------------------
// async copy: 16B per lane, global -> LDS, tracked by ASYNCcnt
// ---------------------------------------------------------------------------
__device__ __forceinline__ void async_copy16(unsigned lds_off, const void* g) {
  asm volatile("global_load_async_to_lds_b128 %0, %1, off"
               :: "v"(lds_off), "v"(g) : "memory");
}
__device__ __forceinline__ void wait_async0() {
  asm volatile("s_wait_asynccnt 0x0" ::: "memory");
}

// A-layout 16x32 f16 fragment: 16 contiguous halves at koff = hi*16.
__device__ __forceinline__ void load_afrag16(HF16& f, const _Float16* row, int hi) {
  const uint4* p = reinterpret_cast<const uint4*>(row + hi * 16);
  f.q4[0] = p[0];
  f.q4[1] = p[1];
}
// B-layout 32x16 f16 fragment: 16 contiguous halves at koff = hi*16.
__device__ __forceinline__ void load_bfrag16(HF16& f, const _Float16* row, int hi) {
  const uint4* p = reinterpret_cast<const uint4*>(row + hi * 16);
  f.q4[0] = p[0];
  f.q4[1] = p[1];
}
// pack two f32 -> one u32 of f16 pairs (v_cvt_pk_rtz_f16_f32)
__device__ __forceinline__ unsigned pkh(float a, float b) {
  union { fp16x2 h; unsigned u; } c;
  c.h = __builtin_amdgcn_cvt_pkrtz(a, b);
  return c.u;
}
__device__ __forceinline__ v8f wmma_f16(const HF16& a, const HF16& b, v8f c) {
  return __builtin_amdgcn_wmma_f32_16x16x32_f16(false, a.v, false, b.v,
                                                (short)0, c, false, false);
}

// ---------------------------------------------------------------------------
// K0: weight-norm for both 1x1 convs, emit f16 weights.
// ---------------------------------------------------------------------------
__global__ void wnorm_kernel(const float* __restrict__ w_qkv,
                             const float* __restrict__ w_out,
                             _Float16* __restrict__ WqN,
                             _Float16* __restrict__ WoN) {
  int row = blockIdx.x;   // 0..1023
  int t   = threadIdx.x;  // 256 = fan_in
  const float* src;
  _Float16*    dst;
  if (row < 768) { src = w_qkv + (size_t)row * CIN;         dst = WqN + (size_t)row * CIN; }
  else           { src = w_out + (size_t)(row - 768) * CIN; dst = WoN + (size_t)(row - 768) * CIN; }
  float v = src[t];
  __shared__ float red[256];
  red[t] = v * v;
  __syncthreads();
  for (int s = 128; s > 0; s >>= 1) {
    if (t < s) red[t] += red[t + s];
    __syncthreads();
  }
  float n     = sqrtf(red[0]);
  float scale = 1.0f / (n + 16.0f * EPSN);
  dst[t] = (_Float16)(v * scale);
}

__global__ void cvt_x_kernel(const float* __restrict__ x,
                             _Float16* __restrict__ xf, int n) {
  int i = blockIdx.x * blockDim.x + threadIdx.x;
  if (i < n) xf[i] = (_Float16)x[i];
}

// ---------------------------------------------------------------------------
// K1: fused QKV GEMM + pixel-norm + repack.
// Block tile 128(channels) x 64(pixels); wave tile 32x32 for q,k,v =>
// 12 v8f accumulators (96 VGPRs). __launch_bounds__(256,1) gives the
// allocator the full VGPR budget -> no accumulator spills.
// Wave wm owns one head's 32 channels: per-pixel sum-of-squares is 16 own
// values + one shfl_xor(16).
// Writes Qn/Kn [bh][pix][32] f16 (Q scaled for exp2 path) and Vt [bh][32][pix].
// ---------------------------------------------------------------------------
__global__ void __launch_bounds__(256, 1)
qkv_gemm_pnorm_kernel(const _Float16* __restrict__ W,   // [768][256]
                      const _Float16* __restrict__ Xf,  // [2][256][4096]
                      _Float16* __restrict__ Qn,
                      _Float16* __restrict__ Kn,
                      _Float16* __restrict__ Vt) {
  const int b    = blockIdx.z;
  const int mt   = blockIdx.y;  // 0..1 (128 channels each)
  const int nt   = blockIdx.x;  // 0..63 (64 pixels each)
  const int t    = threadIdx.x;
  const int lane = t & 31;
  const int wave = t >> 5;
  const int lm   = lane & 15;
  const int hi   = lane >> 4;
  const int wm   = wave & 3;   // head within m-tile
  const int wn   = wave >> 2;  // 0..1 -> n offset wn*32

  __shared__ _Float16 As[3][128 * 40];
  __shared__ _Float16 Bs[64 * 40];

  const _Float16* Bbase = Xf + (size_t)b * CIN * PIX + (size_t)nt * 64;

  v8f acc[3][2][2] = {};

  for (int kb = 0; kb < CIN; kb += 32) {
    {
      int m  = t >> 1;
      int ks = (t & 1) * 16;
#pragma unroll
      for (int m3 = 0; m3 < 3; ++m3) {
        const _Float16* g = W + (size_t)(m3 * 256 + mt * 128 + m) * CIN + kb + ks;
        _Float16*       s = &As[m3][m * 40 + ks];
#pragma unroll
        for (int i = 0; i < 16; i += 8)
          *reinterpret_cast<uint4*>(s + i) = *reinterpret_cast<const uint4*>(g + i);
      }
    }
    {
      int c = t >> 3;          // 0..31
      int p = (t & 7) * 8;     // 0..56
      const _Float16* g = Bbase + (size_t)(kb + c) * PIX + p;
#pragma unroll
      for (int i = 0; i < 8; ++i) Bs[(p + i) * 40 + c] = g[i];
    }
    __syncthreads();

    HF16 bf[2];
#pragma unroll
    for (int j = 0; j < 2; ++j)
      load_bfrag16(bf[j], Bs + (wn * 32 + j * 16 + lm) * 40, hi);
#pragma unroll
    for (int m3 = 0; m3 < 3; ++m3) {
      HF16 af[2];
#pragma unroll
      for (int i = 0; i < 2; ++i)
        load_afrag16(af[i], As[m3] + (wm * 32 + i * 16 + lm) * 40, hi);
#pragma unroll
      for (int i = 0; i < 2; ++i)
#pragma unroll
        for (int j = 0; j < 2; ++j)
          acc[m3][i][j] = wmma_f16(af[i], bf[j], acc[m3][i][j]);
    }
    __syncthreads();
  }

  // fused pixel-norm epilogue; wave owns head = mt*4+wm entirely
  const int bh = b * 8 + (mt * 4 + wm);
#pragma unroll
  for (int j = 0; j < 2; ++j) {
    int pix = nt * 64 + wn * 32 + j * 16 + lm;
    float scl[3];
#pragma unroll
    for (int m3 = 0; m3 < 3; ++m3) {
      float ss = 0.f;
#pragma unroll
      for (int i = 0; i < 2; ++i)
#pragma unroll
        for (int r = 0; r < 8; ++r) { float v = acc[m3][i][j][r]; ss += v * v; }
      ss += __shfl_xor(ss, 16, 32);
      scl[m3] = __builtin_amdgcn_rsqf(ss * (1.0f / HD) + EPSN);
    }
    scl[0] *= QSCALE;

    // Q,K: [pix][d] rows; per i-tile the 8 own d-values are contiguous
#pragma unroll
    for (int i = 0; i < 2; ++i) {
      uint4 wq, wk;
      unsigned* uq = reinterpret_cast<unsigned*>(&wq);
      unsigned* uk = reinterpret_cast<unsigned*>(&wk);
#pragma unroll
      for (int p2 = 0; p2 < 4; ++p2) {
        uq[p2] = pkh(acc[0][i][j][2 * p2] * scl[0], acc[0][i][j][2 * p2 + 1] * scl[0]);
        uk[p2] = pkh(acc[1][i][j][2 * p2] * scl[1], acc[1][i][j][2 * p2 + 1] * scl[1]);
      }
      size_t rowoff = ((size_t)bh * PIX + pix) * HD + i * 16 + 8 * hi;
      *reinterpret_cast<uint4*>(Qn + rowoff) = wq;
      *reinterpret_cast<uint4*>(Kn + rowoff) = wk;
    }
    // V^T: [d][pix] scatter (one-time cost, consumed contiguously by attn)
#pragma unroll
    for (int i = 0; i < 2; ++i)
#pragma unroll
      for (int r = 0; r < 8; ++r) {
        int d = i * 16 + 8 * hi + r;
        Vt[((size_t)bh * HD + d) * PIX + pix] = (_Float16)(acc[2][i][j][r] * scl[2]);
      }
  }
}

// ---------------------------------------------------------------------------
// K2: out GEMM with fused mp_add residual epilogue.
// ---------------------------------------------------------------------------
__global__ void __launch_bounds__(256, 1)
out_gemm_kernel(const _Float16* __restrict__ A,   // WoN [256][256]
                const _Float16* __restrict__ B,   // Yf16 [2][256][4096]
                const float* __restrict__ X,
                float* __restrict__ OUT) {
  const int b    = blockIdx.z;
  const int mt   = blockIdx.y;
  const int nt   = blockIdx.x;
  const int t    = threadIdx.x;
  const int lane = t & 31;
  const int wave = t >> 5;
  const int lm   = lane & 15;
  const int hi   = lane >> 4;
  const int wm   = wave & 3;
  const int wn   = wave >> 2;

  __shared__ _Float16 As[128 * 40];
  __shared__ _Float16 Bs[128 * 40];

  const _Float16* Abase = A + (size_t)mt * 128 * CIN;
  const _Float16* Bbase = B + (size_t)b * CIN * PIX + (size_t)nt * 128;

  v8f acc[2][4] = {};

  for (int kb = 0; kb < CIN; kb += 32) {
    {
      int m  = t >> 1;
      int ks = (t & 1) * 16;
      const _Float16* g = Abase + (size_t)m * CIN + kb + ks;
      _Float16*       s = &As[m * 40 + ks];
#pragma unroll
      for (int i = 0; i < 16; i += 8)
        *reinterpret_cast<uint4*>(s + i) = *reinterpret_cast<const uint4*>(g + i);
    }
    {
      int c = t >> 3;
      int p = (t & 7) * 16;
      const _Float16* g = Bbase + (size_t)(kb + c) * PIX + p;
#pragma unroll
      for (int i = 0; i < 16; ++i) Bs[(p + i) * 40 + c] = g[i];
    }
    __syncthreads();

    HF16 af[2], bf[4];
#pragma unroll
    for (int i = 0; i < 2; ++i)
      load_afrag16(af[i], As + (wm * 32 + i * 16 + lm) * 40, hi);
#pragma unroll
    for (int j = 0; j < 4; ++j)
      load_bfrag16(bf[j], Bs + (wn * 64 + j * 16 + lm) * 40, hi);
#pragma unroll
    for (int i = 0; i < 2; ++i)
#pragma unroll
      for (int j = 0; j < 4; ++j)
        acc[i][j] = wmma_f16(af[i], bf[j], acc[i][j]);
    __syncthreads();
  }

#pragma unroll
  for (int i = 0; i < 2; ++i)
#pragma unroll
    for (int j = 0; j < 4; ++j) {
      int n     = nt * 128 + wn * 64 + j * 16 + lm;
      int mbase = mt * 128 + wm * 32 + i * 16 + hi * 8;
#pragma unroll
      for (int r = 0; r < 8; ++r) {
        size_t idx = ((size_t)b * 256 + mbase + r) * PIX + n;
        const float tm   = 0.3f;
        const float invs = 1.3130643285972254f;
        OUT[idx] = ((1.0f - tm) * X[idx] + tm * acc[i][j][r]) * invs;
      }
    }
}

// ---------------------------------------------------------------------------
// K3: attention, transposed formulation:
//   S^T = K x Q^T  (A = K tile, B = Q^T held in regs all kernel)
//   O^T = V^T x P^T (A = V^T tile, B = P^T built via one half-wave swap)
// Fixed-cap softmax (bounded cosine logits) -> no running max, scalar lsum.
// K/V chunks (128 keys) double-buffered via async-to-LDS.
// ---------------------------------------------------------------------------
__global__ void attn_kernel(const _Float16* __restrict__ Qn,
                            const _Float16* __restrict__ Kn,
                            const _Float16* __restrict__ Vt,
                            _Float16* __restrict__ Y) {
  const int qt   = blockIdx.x;  // 0..31
  const int bh   = blockIdx.y;  // 0..15
  const int b    = bh >> 3, h = bh & 7;
  const int t    = threadIdx.x;
  const int lane = t & 31;
  const int wave = t >> 5;
  const int lm   = lane & 15;
  const int hi   = lane >> 4;

  __shared__ _Float16 Ks[2][128 * 32];  // [buf][key][d]   8KB each
  __shared__ _Float16 Vs[2][32 * 128];  // [buf][d][key]   8KB each

  const int q0 = qt * 128 + wave * 16;
  const _Float16* Kb = Kn + (size_t)bh * PIX * HD;
  const _Float16* Vb = Vt + (size_t)bh * HD * PIX;

  // Q^T B-fragment (lane = q column, 16 d-halves), loaded once
  HF16 qb;
  load_bfrag16(qb, Qn + ((size_t)bh * PIX + q0 + lm) * HD, hi);

  v8f   o0 = {}, o1 = {};
  float lsum = 0.0f;  // one q column per lane

  auto stage = [&](int chunk, int buf) {
    const char* gk = (const char*)(Kb + (size_t)chunk * 128 * HD);
    unsigned lk = (unsigned)(uintptr_t)(&Ks[buf][0]);
#pragma unroll
    for (int i = 0; i < 2; ++i) {
      unsigned off = (unsigned)(((wave * 2 + i) * 32 + lane) * 16);
      async_copy16(lk + off, gk + off);
    }
    unsigned lv = (unsigned)(uintptr_t)(&Vs[buf][0]);
#pragma unroll
    for (int i = 0; i < 2; ++i) {
      int s = wave * 2 + i;
      int d = 2 * s + hi;  // d row 0..31
      const char* gv = (const char*)(Vb + (size_t)d * PIX + chunk * 128) + lm * 16;
      async_copy16(lv + (unsigned)(d * 256 + lm * 16), gv);
    }
  };

  stage(0, 0);

  for (int chunk = 0; chunk < 32; ++chunk) {
    const int buf = chunk & 1;
    wait_async0();     // this wave's chunk data arrived
    __syncthreads();   // all waves' data arrived; prev compute done
    if (chunk + 1 < 32) stage(chunk + 1, buf ^ 1);  // overlaps compute below

    const _Float16* Kc = Ks[buf];
    const _Float16* Vc = Vs[buf];
#pragma unroll
    for (int sub = 0; sub < 4; ++sub) {
      const int kb = sub * 32;
      HF16 kf0, kf1;
      load_afrag16(kf0, Kc + (kb + lm) * HD, hi);
      load_afrag16(kf1, Kc + (kb + 16 + lm) * HD, hi);
      v8f s0 = {}, s1 = {};
      s0 = wmma_f16(kf0, qb, s0);
      s1 = wmma_f16(kf1, qb, s1);

      // raw v_exp_f32 (inputs in [-16.3, 0], denorm path unnecessary)
      float e0[8], e1[8];
#pragma unroll
      for (int r = 0; r < 8; ++r) {
        e0[r] = __builtin_amdgcn_exp2f(s0[r] - LOGCAP);
        e1[r] = __builtin_amdgcn_exp2f(s1[r] - LOGCAP);
        lsum += e0[r] + e1[r];
      }

      // half-wave swap: S^T accumulator layout -> P^T B-fragment layout
      float lo[8], hig[8];
#pragma unroll
      for (int r = 0; r < 8; ++r) {
        float send = hi ? e0[r] : e1[r];
        float recv = __shfl_xor(send, 16, 32);
        lo[r]  = hi ? recv : e0[r];
        hig[r] = hi ? e1[r] : recv;
      }
      HF16 pf;
#pragma unroll
      for (int i = 0; i < 4; ++i) {
        pf.u[i]     = pkh(lo[2 * i], lo[2 * i + 1]);
        pf.u[4 + i] = pkh(hig[2 * i], hig[2 * i + 1]);
      }

      HF16 vf0, vf1;
      load_afrag16(vf0, Vc + lm * 128 + kb, hi);
      load_afrag16(vf1, Vc + (16 + lm) * 128 + kb, hi);
      o0 = wmma_f16(vf0, pf, o0);
      o1 = wmma_f16(vf1, pf, o1);
    }
  }

  lsum += __shfl_xor(lsum, 16, 32);
  float inv = 1.0f / lsum;

  _Float16* Yb = Y + (size_t)b * 256 * PIX + (q0 + lm);
#pragma unroll
  for (int r = 0; r < 8; ++r) {
    int d0 = r + 8 * hi;
    int d1 = 16 + r + 8 * hi;
    Yb[(size_t)(h * HD + d0) * PIX] = (_Float16)(o0[r] * inv);
    Yb[(size_t)(h * HD + d1) * PIX] = (_Float16)(o1[r] * inv);
  }
}

// ---------------------------------------------------------------------------
// host launch
// ---------------------------------------------------------------------------
extern "C" void kernel_launch(void* const* d_in, const int* in_sizes, int n_in,
                              void* d_out, int out_size, void* d_ws, size_t ws_size,
                              hipStream_t stream) {
  (void)in_sizes; (void)n_in; (void)out_size; (void)ws_size;
  const float* x     = (const float*)d_in[0];
  const float* w_qkv = (const float*)d_in[1];
  const float* w_out = (const float*)d_in[2];
  float*       out   = (float*)d_out;

  char*  ws  = (char*)d_ws;
  size_t off = 0;
  auto alloc = [&](size_t bytes) -> void* {
    void* p = ws + off;
    off += (bytes + 255) & ~(size_t)255;
    return p;
  };

  _Float16* WqN  = (_Float16*)alloc((size_t)768 * CIN * 2);
  _Float16* WoN  = (_Float16*)alloc((size_t)256 * CIN * 2);
  _Float16* Xf16 = (_Float16*)alloc((size_t)2 * CIN * PIX * 2);
  _Float16* Qn   = (_Float16*)alloc((size_t)16 * PIX * HD * 2);
  _Float16* Kn   = (_Float16*)alloc((size_t)16 * PIX * HD * 2);
  _Float16* Vt   = (_Float16*)alloc((size_t)16 * HD * PIX * 2);
  _Float16* Yf16 = (_Float16*)alloc((size_t)2 * 256 * PIX * 2);

  wnorm_kernel<<<1024, 256, 0, stream>>>(w_qkv, w_out, WqN, WoN);

  int nx = 2 * CIN * PIX;
  cvt_x_kernel<<<(nx + 255) / 256, 256, 0, stream>>>(x, Xf16, nx);

  qkv_gemm_pnorm_kernel<<<dim3(PIX / 64, 2, 2), 256, 0, stream>>>(
      WqN, Xf16, Qn, Kn, Vt);

  attn_kernel<<<dim3(PIX / 128, 16), 256, 0, stream>>>(Qn, Kn, Vt, Yf16);

  out_gemm_kernel<<<dim3(PIX / 128, 256 / 128, 2), 256, 0, stream>>>(
      WoN, Yf16, x, out);
}
